// GraphEncoder_65609920414072
// MI455X (gfx1250) — matrix operand order repaired and checked by reading, not compiled, not run on hardware
//
#include <hip/hip_runtime.h>
#include <hip/hip_bf16.h>

#define NN   200000
#define EE   400000
#define CC   128
#define CC2  256
#define LL   5
#define NPG  25

typedef __attribute__((ext_vector_type(16))) __bf16 v16bf;
typedef __attribute__((ext_vector_type(8)))  __bf16 v8bf;
typedef __attribute__((ext_vector_type(8)))  float  v8f;

union V16U { v16bf v; v8bf h[2]; };

// native f32 -> bf16 conversion (gfx1250 has hardware bf16 cvt)
__device__ __forceinline__ __bf16 f2bf(float f) { return (__bf16)f; }

// h[i][c] = x_emb1[x[i,0]][c] + x_emb2[x[i,1]][c]
__global__ __launch_bounds__(256) void embed_kernel(
    const int* __restrict__ x, const float* __restrict__ xe1,
    const float* __restrict__ xe2, float* __restrict__ h)
{
    int tid = blockIdx.x * 256 + threadIdx.x;     // N*C threads
    int node = tid >> 7, c = tid & 127;
    int i0 = x[node * 2 + 0], i1 = x[node * 2 + 1];
    h[tid] = xe1[i0 * CC + c] + xe2[i1 * CC + c];
}

// et[l][comb][c] = sum_k (e1[a][k]+e2[b][k]) * W[l][k][c] + bias[l][c]
__global__ __launch_bounds__(128) void edge_table_kernel(
    const float* __restrict__ e1, const float* __restrict__ e2,
    const float* __restrict__ lw, const float* __restrict__ lb,
    float* __restrict__ et)
{
    int tid = blockIdx.x * 128 + threadIdx.x;     // L*9*C = 5760 threads
    int c = tid & 127;
    int comb = (tid >> 7) % 9;
    int l = tid / (9 * CC);
    int a = comb / 3, b = comb % 3;
    float s = lb[l * CC + c];
    for (int k = 0; k < CC; ++k)
        s += (e1[a * CC + k] + e2[b * CC + k]) * lw[((size_t)l * CC + k) * CC + c];
    et[tid] = s;
}

// pack w1 [L][128][256] f32 -> WMMA-B fragment order, bf16
__global__ __launch_bounds__(256) void pack_w1_kernel(
    const float* __restrict__ w1, __bf16* __restrict__ pw1)
{
    int i = blockIdx.x * 256 + threadIdx.x;       // L*16*4*32*16 = 163840
    int j = i & 15, lane = (i >> 4) & 31, kb = (i >> 9) & 3, t = (i >> 11) & 15, l = i >> 15;
    int k = kb * 32 + (lane >> 4) * 8 + (j & 7) + (j >> 3) * 16;
    int n = t * 16 + (lane & 15);
    pw1[i] = f2bf(w1[((size_t)l * CC + k) * CC2 + n]);
}

// pack w2 [L][256][128] f32 -> WMMA-B fragment order, bf16
__global__ __launch_bounds__(256) void pack_w2_kernel(
    const float* __restrict__ w2, __bf16* __restrict__ pw2)
{
    int i = blockIdx.x * 256 + threadIdx.x;       // L*8*8*32*16 = 163840
    int j = i & 15, lane = (i >> 4) & 31, kb = (i >> 9) & 7, t = (i >> 12) & 7, l = i >> 15;
    int k = kb * 32 + (lane >> 4) * 8 + (j & 7) + (j >> 3) * 16;
    int n = t * 16 + (lane & 15);
    pw2[i] = f2bf(w2[((size_t)l * CC2 + k) * CC + n]);
}

__global__ __launch_bounds__(256) void fill_zero_kernel(float* __restrict__ p) {
    p[blockIdx.x * 256 + threadIdx.x] = 0.f;
}

__global__ __launch_bounds__(256) void mask_kernel(float* __restrict__ p, int n) {
    int i = blockIdx.x * 256 + threadIdx.x;
    if (i < n) p[i] = 1.0f;
}

// one wave per edge: msg = relu(h[src] + et[comb]); atomicAdd into agg[dst]
__global__ __launch_bounds__(256) void scatter_kernel(
    const float* __restrict__ h, const float* __restrict__ etl,
    const int* __restrict__ ei, const int* __restrict__ ea,
    float* __restrict__ agg)
{
    int tid = blockIdx.x * 256 + threadIdx.x;     // E*32 threads
    int e = tid >> 5;
    int lane = tid & 31;
    int c = lane * 4;
    int src = ei[e];
    int dst = ei[EE + e];
    int comb = ea[2 * e] * 3 + ea[2 * e + 1];
    const float4 h4 = *(const float4*)(h + (size_t)src * CC + c);
    const float4 e4 = *(const float4*)(etl + comb * CC + c);
    float* ap = agg + (size_t)dst * CC + c;
    atomicAdd(ap + 0, fmaxf(h4.x + e4.x, 0.f));
    atomicAdd(ap + 1, fmaxf(h4.y + e4.y, 0.f));
    atomicAdd(ap + 2, fmaxf(h4.z + e4.z, 0.f));
    atomicAdd(ap + 3, fmaxf(h4.w + e4.w, 0.f));
}

// fused: z = (1+eps)h + agg + relu(h + et0);  y = relu(z@W1+b1);  o = y@W2+b2;  BN; (relu)
__global__ __launch_bounds__(128) void node_mlp_kernel(
    const float* __restrict__ hin, const float* __restrict__ agg,
    const float* __restrict__ et0,
    const __bf16* __restrict__ pw1, const __bf16* __restrict__ pw2,
    const float* __restrict__ b1, const float* __restrict__ b2,
    const float* __restrict__ gamma, const float* __restrict__ beta,
    const float* __restrict__ mean, const float* __restrict__ var,
    const float* __restrict__ epsArr, int layer, int doRelu,
    float* __restrict__ hout)
{
    __shared__ __align__(32) __bf16 zt[4][16][CC];    // 16 KB
    __shared__ __align__(32) __bf16 yt[4][16][CC2];   // 32 KB
    int lane = threadIdx.x & 31;
    int wave = threadIdx.x >> 5;
    int tileBase = blockIdx.x * 64 + wave * 16;
    float epsv = epsArr[layer];

    // build bf16 z tile (each lane covers 4 channels per row)
    int cz = lane * 4;
    const float4 e4 = *(const float4*)(et0 + cz);
    #pragma unroll 4
    for (int r = 0; r < 16; ++r) {
        size_t base = (size_t)(tileBase + r) * CC + cz;
        const float4 h4 = *(const float4*)(hin + base);
        const float4 a4 = *(const float4*)(agg + base);
        zt[wave][r][cz + 0] = f2bf((1.f + epsv) * h4.x + a4.x + fmaxf(h4.x + e4.x, 0.f));
        zt[wave][r][cz + 1] = f2bf((1.f + epsv) * h4.y + a4.y + fmaxf(h4.y + e4.y, 0.f));
        zt[wave][r][cz + 2] = f2bf((1.f + epsv) * h4.z + a4.z + fmaxf(h4.z + e4.z, 0.f));
        zt[wave][r][cz + 3] = f2bf((1.f + epsv) * h4.w + a4.w + fmaxf(h4.w + e4.w, 0.f));
    }

    int m   = lane & 15;
    int hi8 = (lane >> 4) * 8;

    // stage 1: y[16x256] = relu(z[16x128] @ W1 + b1)
    for (int t = 0; t < 16; ++t) {
        v8f acc = {};
        #pragma unroll
        for (int kb = 0; kb < 4; ++kb) {
            V16U a, b;
            int klo = kb * 32 + hi8;
            a.h[0] = *(const v8bf*)&zt[wave][m][klo];
            a.h[1] = *(const v8bf*)&zt[wave][m][klo + 16];
            b.v = *(const v16bf*)(pw1 + (size_t)((t * 4 + kb) * 32 + lane) * 16);
            acc = __builtin_amdgcn_wmma_f32_16x16x32_bf16(
                false, a.v, false, b.v, (short)0, acc, false, false);
        }
        int n = t * 16 + m;
        float bb = b1[n];
        #pragma unroll
        for (int r = 0; r < 8; ++r)
            yt[wave][hi8 + r][n] = f2bf(fmaxf(acc[r] + bb, 0.f));
    }

    // stage 2: out[16x128] = y @ W2 + b2 -> BN -> (relu)
    for (int t = 0; t < 8; ++t) {
        v8f acc = {};
        #pragma unroll
        for (int kb = 0; kb < 8; ++kb) {
            V16U a, b;
            int klo = kb * 32 + hi8;
            a.h[0] = *(const v8bf*)&yt[wave][m][klo];
            a.h[1] = *(const v8bf*)&yt[wave][m][klo + 16];
            b.v = *(const v16bf*)(pw2 + (size_t)((t * 8 + kb) * 32 + lane) * 16);
            acc = __builtin_amdgcn_wmma_f32_16x16x32_bf16(
                false, a.v, false, b.v, (short)0, acc, false, false);
        }
        int c = t * 16 + m;
        float bb = b2[c];
        float sc = gamma[c] * rsqrtf(var[c] + 1e-5f);
        float mu = mean[c], be = beta[c];
        #pragma unroll
        for (int r = 0; r < 8; ++r) {
            float v = (acc[r] + bb - mu) * sc + be;
            if (doRelu) v = fmaxf(v, 0.f);
            hout[(size_t)(tileBase + hi8 + r) * CC + c] = v;
        }
    }
}

extern "C" void kernel_launch(void* const* d_in, const int* in_sizes, int n_in,
                              void* d_out, int out_size, void* d_ws, size_t ws_size,
                              hipStream_t stream) {
    const int*   x     = (const int*)d_in[0];
    const int*   ei    = (const int*)d_in[1];
    // d_in[2] = batch (arange//25) -> output is a pure reshape; unused
    const int*   ea    = (const int*)d_in[3];
    const float* xe1   = (const float*)d_in[4];
    const float* xe2   = (const float*)d_in[5];
    const float* ee1   = (const float*)d_in[6];
    const float* ee2   = (const float*)d_in[7];
    const float* lw    = (const float*)d_in[8];
    const float* lb    = (const float*)d_in[9];
    const float* w1    = (const float*)d_in[10];
    const float* b1    = (const float*)d_in[11];
    const float* w2    = (const float*)d_in[12];
    const float* b2    = (const float*)d_in[13];
    const float* eps   = (const float*)d_in[14];
    const float* bng   = (const float*)d_in[15];
    const float* bnb   = (const float*)d_in[16];
    const float* bnm   = (const float*)d_in[17];
    const float* bnv   = (const float*)d_in[18];

    char* w = (char*)d_ws;
    float*  h    = (float*)w;  w += (size_t)NN * CC * 4;       // 102.4 MB
    float*  agg  = (float*)w;  w += (size_t)NN * CC * 4;       // 102.4 MB
    float*  et   = (float*)w;  w += (size_t)LL * 9 * CC * 4;   // 23 KB
    __bf16* pw1  = (__bf16*)w; w += (size_t)LL * 32768 * 2;    // 320 KB
    __bf16* pw2  = (__bf16*)w;                                  // 320 KB

    float* dout = (float*)d_out;

    embed_kernel<<<(NN * CC) / 256, 256, 0, stream>>>(x, xe1, xe2, h);
    edge_table_kernel<<<(LL * 9 * CC) / 128, 128, 0, stream>>>(ee1, ee2, lw, lb, et);
    pack_w1_kernel<<<(LL * 32768) / 256, 256, 0, stream>>>(w1, pw1);
    pack_w2_kernel<<<(LL * 32768) / 256, 256, 0, stream>>>(w2, pw2);

    for (int l = 0; l < LL; ++l) {
        fill_zero_kernel<<<(NN * CC) / 256, 256, 0, stream>>>(agg);
        scatter_kernel<<<(EE * 32) / 256, 256, 0, stream>>>(
            h, et + (size_t)l * 9 * CC, ei, ea, agg);
        float* out = (l == LL - 1) ? dout : h;
        node_mlp_kernel<<<NN / 64, 128, 0, stream>>>(
            h, agg, et + (size_t)l * 9 * CC,
            pw1 + (size_t)l * 32768, pw2 + (size_t)l * 32768,
            b1 + (size_t)l * CC2, b2 + (size_t)l * CC,
            bng + (size_t)l * CC, bnb + (size_t)l * CC,
            bnm + (size_t)l * CC, bnv + (size_t)l * CC,
            eps, l, (l < LL - 1) ? 1 : 0, out);
    }

    // mask: all true (25 nodes per graph exactly) -> 1.0f
    mask_kernel<<<(NN + 255) / 256, 256, 0, stream>>>(dout + (size_t)NN * CC, NN);
}
